// Model_16655883174368
// MI455X (gfx1250) — matrix operand order repaired and checked by reading, not compiled, not run on hardware
//
#include <hip/hip_runtime.h>
#include <hip/hip_bf16.h>
#include <math.h>

// ---------------------------------------------------------------------------
// MI455X (gfx1250). wave32; bf16 WMMA 16x16x32 for all dense GEMMs with
// N-major (pre-transposed) bf16 weights so every LDS fragment access is a
// contiguous ds_load_b128; Tensor Data Mover (tensor_load_to_lds + TENSORcnt)
// feeds all GEMM LDS tiles with hardware padding to the conflict-free stride;
// VALU for the tiny ProbSparse part; split-K WMMA + f32 atomics for the
// bandwidth-bound final projection.
// ---------------------------------------------------------------------------

typedef __bf16 bf16;
typedef __attribute__((ext_vector_type(16))) __bf16    v16bf;
typedef __attribute__((ext_vector_type(8)))  float     v8f;
typedef __attribute__((ext_vector_type(4)))  unsigned  v4u;
typedef __attribute__((ext_vector_type(8)))  int       v8i;
typedef __attribute__((ext_vector_type(4)))  int       v4i;

#define LSEQ   2048
#define BATCH  16
#define DMODEL 128
#define NH     8
#define DH     16
#define FFN    512
#define NSAMP  40
#define MNROWS (BATCH * LSEQ)      // 32768
#define PROJK  (LSEQ * DMODEL)     // 262144

#if defined(__gfx1250__) && __has_builtin(__builtin_amdgcn_tensor_load_to_lds) && \
    __has_builtin(__builtin_amdgcn_s_wait_tensorcnt)
#define HAVE_TDM 1
#else
#define HAVE_TDM 0
#endif

static __device__ __forceinline__ float gelu_exact(float v) {
    return 0.5f * v * (1.0f + erff(v * 0.70710678118654752f));
}

union frag_u { uint4 q[2]; v16bf v; };

#if HAVE_TDM
// Issue one TDM 2D tile load: 32 (K) x rows tile of bf16, global row stride
// row_stride_elems, into LDS at lds_byte_off with +4dw pad every 16dw so the
// landed row stride is 80B == the bf16[..][40] layout the fragment loads use.
static __device__ __forceinline__ void tdm_load_tile(const bf16* gsrc,
                                                     unsigned lds_byte_off,
                                                     unsigned rows,
                                                     unsigned row_stride_elems) {
    const unsigned long long ga = (unsigned long long)(size_t)gsrc;
    // D# group0 (ISA 8.3): count=1 | lds_addr | global_addr(57b) | type=2
    v4u g0 = { 1u,
               lds_byte_off,
               (unsigned)ga,
               (unsigned)((ga >> 32) & 0x01FFFFFFull) | (2u << 30) };
    // D# group1 (ISA 8.4): mask=0, data_size=1(2B), pad_enable=1,
    // pad_interval=3(16dw), pad_amount=3(4dw); dims/strides in elements.
    const unsigned w0   = (1u << 16) | (1u << 20) | (3u << 22) | (3u << 25);
    const unsigned dim0 = row_stride_elems;     // tensor width
    const unsigned dim1 = rows;                 // tensor height
    v8i g1 = { (int)w0,
               (int)((dim0 & 0xFFFFu) << 16),                       // dim0[15:0]
               (int)((dim0 >> 16) | ((dim1 & 0xFFFFu) << 16)),      // dim0 hi | dim1 lo
               (int)((dim1 >> 16) | (32u << 16)),                   // dim1 hi | tile_dim0=32
               (int)rows,                                           // tile_dim1 (tile_dim2=0)
               (int)row_stride_elems,                               // dim0_stride[31:0]
               0, 0 };
    v4i z4 = {0, 0, 0, 0};
#if __clang_major__ >= 23
    v8i z8 = {0, 0, 0, 0, 0, 0, 0, 0};
    __builtin_amdgcn_tensor_load_to_lds(g0, g1, z4, z4, z8, 0);
#else
    __builtin_amdgcn_tensor_load_to_lds(g0, g1, z4, z4, 0);
#endif
}
static __device__ __forceinline__ unsigned lds_addr32(const void* p) {
    return (unsigned)(size_t)p;   // flat LDS address: low 32 bits = LDS offset
}
#endif

// -------------------------------- embed ------------------------------------
// tok = 3-tap circular conv over L (C=16 -> D=128) + sinusoidal PE.
__global__ __launch_bounds__(128) void k_embed(const float* __restrict__ xe,
                                               const float* __restrict__ cw,
                                               float* __restrict__ X,
                                               bf16*  __restrict__ Xb) {
    const int bl = blockIdx.x;            // b*L + l
    const int l  = bl & (LSEQ - 1);
    const int b  = bl >> 11;
    const int d  = threadIdx.x;
    __shared__ float sx[3][16];
    if (d < 48) {
        const int k = d >> 4, c = d & 15;
        const int lp = (l + k - 1 + LSEQ) & (LSEQ - 1);
        sx[k][c] = xe[((size_t)b * LSEQ + lp) * 16 + c];
    }
    __syncthreads();
    float acc = 0.0f;
    #pragma unroll
    for (int k = 0; k < 3; ++k) {
        const float* w = cw + (size_t)k * 16 * DMODEL + d;   // cw[k][c][d]
        #pragma unroll
        for (int c = 0; c < 16; ++c) acc += sx[k][c] * w[c * DMODEL];
    }
    const int j = d >> 1;
    const float div = expf((float)(2 * j) * (-9.210340371976184f / (float)DMODEL));
    const float ang = (float)l * div;
    acc += (d & 1) ? cosf(ang) : sinf(ang);
    const size_t o = (size_t)bl * DMODEL + d;
    X[o]  = acc;
    Xb[o] = (bf16)acc;
}

// ----------------- fp32 (K,N) -> bf16 (N,K) tiled transpose ----------------
__global__ __launch_bounds__(256) void k_f2bf_t(const float* __restrict__ in,
                                                bf16* __restrict__ outp,
                                                int K, int N) {
    __shared__ bf16 tile[32][33];
    const int k0 = blockIdx.x * 32;
    const int n0 = blockIdx.y * 32;
    const int tn = threadIdx.x & 31, tk = threadIdx.x >> 5;   // 32 x 8
    #pragma unroll
    for (int i = 0; i < 32; i += 8)
        tile[tk + i][tn] = (bf16)in[(size_t)(k0 + tk + i) * N + n0 + tn];
    __syncthreads();
    #pragma unroll
    for (int i = 0; i < 32; i += 8)
        outp[(size_t)(n0 + tk + i) * K + k0 + tn] = tile[tn][tk + i];
}

// ------------------------------- WMMA GEMM ---------------------------------
// C(M,N) = A(M,K) @ B(K,N) + bias [, gelu];  B passed N-major as Bt(N,K).
// 256 thr = 8 waves; block tile 128x64; K-step 32; double-buffered LDS fed by
// the Tensor Data Mover (wave0: A tile, wave1: B tile), overlapped via
// s_wait_tensorcnt(1). Fragment reads are conflict-free ds_load_b128.
__global__ __launch_bounds__(256) void k_gemm(const bf16* __restrict__ A,
                                              const bf16* __restrict__ Bt,
                                              const float* __restrict__ bias,
                                              float* __restrict__ C,
                                              bf16*  __restrict__ Cb,
                                              int M, int N, int K, int do_gelu) {
    __shared__ bf16 As[2][128][40];   // landed row = 80B (TDM pad), 16B-aligned
    __shared__ bf16 Bs[2][64][40];    // N-major B tile: Bs[.][n_local][k_local]
    const int t    = threadIdx.x;
    const int lane = t & 31;
    const int wv   = t >> 5;
    const int m0   = blockIdx.x * 128;
    const int n0   = blockIdx.y * 64;

    v8f acc0 = {0,0,0,0,0,0,0,0};
    v8f acc1 = acc0, acc2 = acc0, acc3 = acc0;

    const int nl = lane & 15;
    const int ml = 16 * wv + nl;
    const int ka = (lane < 16) ? 0 : 16;        // A frag K split (contiguous k = 16h + s, same order as B)
    const int kb = (lane < 16) ? 0 : 16;        // B frag K split
    int cur = 0;

#if HAVE_TDM
    const unsigned offA[2] = { lds_addr32(&As[0][0][0]), lds_addr32(&As[1][0][0]) };
    const unsigned offB[2] = { lds_addr32(&Bs[0][0][0]), lds_addr32(&Bs[1][0][0]) };
    if (wv == 0) tdm_load_tile(A  + (size_t)m0 * K, offA[0], 128, (unsigned)K);
    if (wv == 1) tdm_load_tile(Bt + (size_t)n0 * K, offB[0],  64, (unsigned)K);
#else
    const int ar = t >> 1, ac = (t & 1) * 16;   // A tile: 128x32, 16 elems/thr
    const int br = t >> 2, bc = (t & 3) * 8;    // B tile:  64x32,  8 elems/thr
#endif

    for (int k0 = 0; k0 < K; k0 += 32) {
#if HAVE_TDM
        const int more = (k0 + 32 < K);
        if (more) {
            if (wv == 0) tdm_load_tile(A  + (size_t)m0 * K + k0 + 32, offA[cur ^ 1], 128, (unsigned)K);
            if (wv == 1) tdm_load_tile(Bt + (size_t)n0 * K + k0 + 32, offB[cur ^ 1],  64, (unsigned)K);
        }
        if (wv < 2) {
            if (more) __builtin_amdgcn_s_wait_tensorcnt(1);
            else      __builtin_amdgcn_s_wait_tensorcnt(0);
        }
        __syncthreads();
#else
        const uint4* ag = (const uint4*)(A + (size_t)(m0 + ar) * K + k0 + ac);
        uint4 a0 = ag[0], a1 = ag[1];
        uint4 b0 = *((const uint4*)(Bt + (size_t)(n0 + br) * K + k0 + bc));
        if (k0 + 32 < K)
            __builtin_prefetch(A + (size_t)(m0 + ar) * K + (k0 + 32) + ac, 0, 1);
        __syncthreads();
        *((uint4*)&As[0][ar][ac])     = a0;
        *((uint4*)&As[0][ar][ac + 8]) = a1;
        *((uint4*)&Bs[0][br][bc])     = b0;
        __syncthreads();
#endif

        frag_u au, bu0, bu1, bu2, bu3;
        au.q[0]  = *((const uint4*)&As[cur][ml][ka]);
        au.q[1]  = *((const uint4*)&As[cur][ml][ka + 8]);
        bu0.q[0] = *((const uint4*)&Bs[cur][nl][kb]);
        bu0.q[1] = *((const uint4*)&Bs[cur][nl][kb + 8]);
        bu1.q[0] = *((const uint4*)&Bs[cur][16 + nl][kb]);
        bu1.q[1] = *((const uint4*)&Bs[cur][16 + nl][kb + 8]);
        bu2.q[0] = *((const uint4*)&Bs[cur][32 + nl][kb]);
        bu2.q[1] = *((const uint4*)&Bs[cur][32 + nl][kb + 8]);
        bu3.q[0] = *((const uint4*)&Bs[cur][48 + nl][kb]);
        bu3.q[1] = *((const uint4*)&Bs[cur][48 + nl][kb + 8]);

        acc0 = __builtin_amdgcn_wmma_f32_16x16x32_bf16(false, au.v, false, bu0.v, (short)0, acc0, false, false);
        acc1 = __builtin_amdgcn_wmma_f32_16x16x32_bf16(false, au.v, false, bu1.v, (short)0, acc1, false, false);
        acc2 = __builtin_amdgcn_wmma_f32_16x16x32_bf16(false, au.v, false, bu2.v, (short)0, acc2, false, false);
        acc3 = __builtin_amdgcn_wmma_f32_16x16x32_bf16(false, au.v, false, bu3.v, (short)0, acc3, false, false);

#if HAVE_TDM
        __syncthreads();      // buffer cur free for the TDM issue in step k0+64
        cur ^= 1;
#endif
    }

    // C/D layout: element r -> row r (lanes 0-15) / r+8 (lanes 16-31); col = lane&15.
    const int rbase = (lane >> 4) << 3;
#define EPILOGUE(ACC, NT)                                                     \
    {                                                                         \
        const int col  = n0 + 16 * (NT) + nl;                                 \
        const float bb = bias[col];                                           \
        _Pragma("unroll")                                                     \
        for (int r = 0; r < 8; ++r) {                                         \
            const int row = m0 + 16 * wv + rbase + r;                         \
            float v = ACC[r] + bb;                                            \
            if (do_gelu) v = gelu_exact(v);                                   \
            const size_t o = (size_t)row * N + col;                           \
            C[o] = v;                                                         \
            if (Cb) Cb[o] = (bf16)v;                                          \
        }                                                                     \
    }
    EPILOGUE(acc0, 0)
    EPILOGUE(acc1, 1)
    EPILOGUE(acc2, 2)
    EPILOGUE(acc3, 3)
#undef EPILOGUE
}

// ----------------------- ProbSparse: sampled scores ------------------------
__global__ __launch_bounds__(256) void k_scores(const float* __restrict__ Q,
                                                const float* __restrict__ Kf,
                                                float* __restrict__ Mout) {
    const int gid = blockIdx.x * 256 + threadIdx.x;   // over B*H*L
    const int l  = gid & (LSEQ - 1);
    const int bh = gid >> 11;
    const int b  = bh >> 3, h = bh & 7;
    const float* qv = Q + ((size_t)(b * LSEQ + l) * NH + h) * DH;
    float qr[DH];
    #pragma unroll
    for (int d = 0; d < DH; ++d) qr[d] = qv[d];
    float mx = -1e30f, sm = 0.0f;
    for (int u = 0; u < NSAMP; ++u) {
        unsigned s = (unsigned)(l * NSAMP + u) * 2654435761u + 0x9E3779B9u;
        s ^= s >> 16; s *= 2246822519u; s ^= s >> 13;
        const int kidx = (int)(s & (LSEQ - 1));
        const float* kv = Kf + ((size_t)(b * LSEQ + kidx) * NH + h) * DH;
        float a = 0.0f;
        #pragma unroll
        for (int d = 0; d < DH; ++d) a += qr[d] * kv[d];
        mx = fmaxf(mx, a); sm += a;
    }
    Mout[(size_t)bh * LSEQ + l] = mx - sm * (1.0f / NSAMP);
}

// ------------------------------- top-k (40) --------------------------------
__global__ __launch_bounds__(256) void k_topk(const float* __restrict__ M,
                                              int* __restrict__ topk) {
    const int bh = blockIdx.x, t = threadIdx.x;
    __shared__ float sM[LSEQ];
    __shared__ float rv[256];
    __shared__ int   ri[256];
    for (int i = t; i < LSEQ; i += 256) sM[i] = M[(size_t)bh * LSEQ + i];
    __syncthreads();
    for (int it = 0; it < NSAMP; ++it) {
        float bv = -1e30f; int bi = 0;
        for (int i = t; i < LSEQ; i += 256)
            if (sM[i] > bv) { bv = sM[i]; bi = i; }
        rv[t] = bv; ri[t] = bi; __syncthreads();
        for (int s = 128; s > 0; s >>= 1) {
            if (t < s && rv[t + s] > rv[t]) { rv[t] = rv[t + s]; ri[t] = ri[t + s]; }
            __syncthreads();
        }
        if (t == 0) { topk[bh * NSAMP + it] = ri[0]; sM[ri[0]] = -1e30f; }
        __syncthreads();
    }
}

// --------------------- ctx = broadcast(mean(V, axis=L)) --------------------
__global__ __launch_bounds__(256) void k_ctx(const float* __restrict__ V,
                                             bf16* __restrict__ attnb) {
    const int bh = blockIdx.x, t = threadIdx.x;
    const int b = bh >> 3, h = bh & 7;
    const int d = t & 15, grp = t >> 4;
    __shared__ float red[16][17];
    __shared__ float vm[16];
    float s = 0.0f;
    for (int l = grp; l < LSEQ; l += 16)
        s += V[((size_t)(b * LSEQ + l) * NH + h) * DH + d];
    red[grp][d] = s; __syncthreads();
    if (t < 16) {
        float a = 0.0f;
        for (int g = 0; g < 16; ++g) a += red[g][t];
        vm[t] = a * (1.0f / LSEQ);
    }
    __syncthreads();
    for (int l = t; l < LSEQ; l += 256) {
        const size_t o = (size_t)(b * LSEQ + l) * DMODEL + h * DH;
        #pragma unroll
        for (int d2 = 0; d2 < DH; ++d2) attnb[o + d2] = (bf16)vm[d2];
    }
}

// ----------- full softmax attention for the 40 selected queries ------------
__global__ __launch_bounds__(256) void k_topq(const float* __restrict__ Q,
                                              const float* __restrict__ Kf,
                                              const float* __restrict__ V,
                                              const int* __restrict__ topk,
                                              bf16* __restrict__ attnb) {
    const int id = blockIdx.x;       // bh*NSAMP + u
    const int u  = id % NSAMP;
    const int bh = id / NSAMP;
    const int b  = bh >> 3, h = bh & 7;
    const int t  = threadIdx.x;
    const int lq = topk[bh * NSAMP + u];
    __shared__ float qs[DH];
    __shared__ float red[256];
    __shared__ float outv[DH];
    if (t < DH) qs[t] = Q[((size_t)(b * LSEQ + lq) * NH + h) * DH + t];
    __syncthreads();
    float sc[8], mx = -1e30f;
    #pragma unroll
    for (int j = 0; j < 8; ++j) {
        const int kk = t + j * 256;
        const float* kv = Kf + ((size_t)(b * LSEQ + kk) * NH + h) * DH;
        float a = 0.0f;
        #pragma unroll
        for (int d = 0; d < DH; ++d) a += qs[d] * kv[d];
        sc[j] = a * 0.25f;                // 1/sqrt(16)
        mx = fmaxf(mx, sc[j]);
    }
    red[t] = mx; __syncthreads();
    for (int s = 128; s > 0; s >>= 1) {
        if (t < s) red[t] = fmaxf(red[t], red[t + s]);
        __syncthreads();
    }
    const float gmx = red[0]; __syncthreads();
    float ps = 0.0f, pv[DH];
    #pragma unroll
    for (int d = 0; d < DH; ++d) pv[d] = 0.0f;
    #pragma unroll
    for (int j = 0; j < 8; ++j) {
        const int kk = t + j * 256;
        const float p = expf(sc[j] - gmx);
        ps += p;
        const float* vv = V + ((size_t)(b * LSEQ + kk) * NH + h) * DH;
        #pragma unroll
        for (int d = 0; d < DH; ++d) pv[d] += p * vv[d];
    }
    red[t] = ps; __syncthreads();
    for (int s = 128; s > 0; s >>= 1) {
        if (t < s) red[t] += red[t + s];
        __syncthreads();
    }
    const float gs = red[0];
    if (t < DH) outv[t] = 0.0f;
    __syncthreads();
    #pragma unroll
    for (int d = 0; d < DH; ++d) atomicAdd(&outv[d], pv[d]);
    __syncthreads();
    if (t < DH) {
        const size_t o = (size_t)(b * LSEQ + lq) * DMODEL + h * DH + t;
        attnb[o] = (bf16)(outv[t] / gs);
    }
}

// --------------------- residual + LayerNorm (+ GELU) -----------------------
__global__ __launch_bounds__(128) void k_add_ln(const float* __restrict__ X,
                                                const float* __restrict__ A,
                                                const float* __restrict__ g,
                                                const float* __restrict__ bta,
                                                float* __restrict__ Yf,
                                                bf16*  __restrict__ Yb,
                                                int do_gelu) {
    const int row = blockIdx.x, t = threadIdx.x;
    __shared__ float red[128];
    const size_t o = (size_t)row * DMODEL + t;
    float v = X[o] + (A ? A[o] : 0.0f);
    red[t] = v; __syncthreads();
    for (int s = 64; s > 0; s >>= 1) { if (t < s) red[t] += red[t + s]; __syncthreads(); }
    const float mean = red[0] * (1.0f / DMODEL); __syncthreads();
    const float d0 = v - mean;
    red[t] = d0 * d0; __syncthreads();
    for (int s = 64; s > 0; s >>= 1) { if (t < s) red[t] += red[t + s]; __syncthreads(); }
    const float var = red[0] * (1.0f / DMODEL);
    float y = d0 * rsqrtf(var + 1e-5f) * g[t] + bta[t];
    if (do_gelu) y = gelu_exact(y);
    if (Yf) Yf[o] = y;
    if (Yb) Yb[o] = (bf16)y;
}

// ----------------------- output init with bias -----------------------------
__global__ void k_out_init(const float* __restrict__ pb, float* __restrict__ outp) {
    const int i = blockIdx.x * 256 + threadIdx.x;   // 2048 total
    outp[i] = pb[i & (DMODEL - 1)];
}

// ------------- final projection: split-K WMMA + f32 atomics ----------------
// out(16,128) += A(16,PROJK) @ W(PROJK,128); W passed N-major as Bt(128,PROJK).
// One block (128 thr, 4 waves) per 2048-wide K chunk; TDM streams the weight
// tile (wave0) and activation tile (wave1) while the waves run WMMA.
__global__ __launch_bounds__(128) void k_proj(const bf16* __restrict__ A,
                                              const bf16* __restrict__ Bt,
                                              float* __restrict__ outp) {
    __shared__ bf16 As[2][16][40];
    __shared__ bf16 Bs[2][128][40];
    const int t = threadIdx.x, lane = t & 31, wv = t >> 5;
    const int kbase = blockIdx.x * 2048;
    v8f acc0 = {0,0,0,0,0,0,0,0};
    v8f acc1 = acc0;
    const int nl = lane & 15;
    const int ka = (lane < 16) ? 0 : 16;
    const int kb = (lane < 16) ? 0 : 16;
    const int c0 = 32 * wv + nl;
    int cur = 0;

#if HAVE_TDM
    const unsigned offA[2] = { lds_addr32(&As[0][0][0]), lds_addr32(&As[1][0][0]) };
    const unsigned offB[2] = { lds_addr32(&Bs[0][0][0]), lds_addr32(&Bs[1][0][0]) };
    if (wv == 0) tdm_load_tile(Bt + kbase, offB[0], 128, PROJK);
    if (wv == 1) tdm_load_tile(A  + kbase, offA[0],  16, PROJK);
#else
    const int ar = t >> 3, ac = (t & 7) * 4;   // A tile 16x32, uint2 per thread
#endif

    for (int kk = 0; kk < 2048; kk += 32) {
#if HAVE_TDM
        const int more = (kk + 32 < 2048);
        if (more) {
            if (wv == 0) tdm_load_tile(Bt + kbase + kk + 32, offB[cur ^ 1], 128, PROJK);
            if (wv == 1) tdm_load_tile(A  + kbase + kk + 32, offA[cur ^ 1],  16, PROJK);
        }
        if (wv < 2) {
            if (more) __builtin_amdgcn_s_wait_tensorcnt(1);
            else      __builtin_amdgcn_s_wait_tensorcnt(0);
        }
        __syncthreads();
#else
        const int k0 = kbase + kk;
        const uint2 av = *((const uint2*)(A + (size_t)ar * PROJK + k0 + ac));
        const uint4* bg = (const uint4*)(Bt + (size_t)t * PROJK + k0);  // row n=t
        uint4 b0 = bg[0], b1 = bg[1], b2 = bg[2], b3 = bg[3];
        __syncthreads();
        *((uint2*)&As[0][ar][ac]) = av;
        *((uint4*)&Bs[0][t][0])   = b0;
        *((uint4*)&Bs[0][t][8])   = b1;
        *((uint4*)&Bs[0][t][16])  = b2;
        *((uint4*)&Bs[0][t][24])  = b3;
        __syncthreads();
#endif
        frag_u au, bu0, bu1;
        au.q[0]  = *((const uint4*)&As[cur][nl][ka]);
        au.q[1]  = *((const uint4*)&As[cur][nl][ka + 8]);
        bu0.q[0] = *((const uint4*)&Bs[cur][c0][kb]);
        bu0.q[1] = *((const uint4*)&Bs[cur][c0][kb + 8]);
        bu1.q[0] = *((const uint4*)&Bs[cur][c0 + 16][kb]);
        bu1.q[1] = *((const uint4*)&Bs[cur][c0 + 16][kb + 8]);
        acc0 = __builtin_amdgcn_wmma_f32_16x16x32_bf16(false, au.v, false, bu0.v, (short)0, acc0, false, false);
        acc1 = __builtin_amdgcn_wmma_f32_16x16x32_bf16(false, au.v, false, bu1.v, (short)0, acc1, false, false);
#if HAVE_TDM
        __syncthreads();
        cur ^= 1;
#endif
    }
    const int rb = (lane >> 4) << 3;
    #pragma unroll
    for (int r = 0; r < 8; ++r) {
        atomicAdd(&outp[(rb + r) * DMODEL + 32 * wv + nl],      acc0[r]);
        atomicAdd(&outp[(rb + r) * DMODEL + 32 * wv + 16 + nl], acc1[r]);
    }
}

// ---------------------------------------------------------------------------
extern "C" void kernel_launch(void* const* d_in, const int* in_sizes, int n_in,
                              void* d_out, int out_size, void* d_ws, size_t ws_size,
                              hipStream_t stream) {
    (void)in_sizes; (void)n_in; (void)out_size; (void)ws_size;
    const float* x_enc  = (const float*)d_in[0];
    const float* conv_W = (const float*)d_in[1];
    const float* Wq = (const float*)d_in[2];  const float* bq = (const float*)d_in[3];
    const float* Wk = (const float*)d_in[4];  const float* bk = (const float*)d_in[5];
    const float* Wv = (const float*)d_in[6];  const float* bv = (const float*)d_in[7];
    const float* Wo = (const float*)d_in[8];  const float* bo = (const float*)d_in[9];
    const float* W1 = (const float*)d_in[10]; const float* b1 = (const float*)d_in[11];
    const float* W2 = (const float*)d_in[12]; const float* b2 = (const float*)d_in[13];
    const float* ln1_g = (const float*)d_in[14]; const float* ln1_b = (const float*)d_in[15];
    const float* ln2_g = (const float*)d_in[16]; const float* ln2_b = (const float*)d_in[17];
    const float* norm_g = (const float*)d_in[18]; const float* norm_b = (const float*)d_in[19];
    const float* proj_W = (const float*)d_in[20]; const float* proj_b = (const float*)d_in[21];
    float* outp = (float*)d_out;

    // ---- workspace bump allocator (256B aligned) ----
    char* base = (char*)d_ws; size_t off = 0;
    auto alloc = [&](size_t bytes) -> void* {
        off = (off + 255) & ~(size_t)255;
        void* p = base + off; off += bytes; return p;
    };
    const size_t MN = MNROWS;
    float* x   = (float*)alloc(MN * DMODEL * 4);
    bf16*  xb  = (bf16*) alloc(MN * DMODEL * 2);
    float* qf  = (float*)alloc(MN * DMODEL * 4);
    float* kf  = (float*)alloc(MN * DMODEL * 4);
    float* vf  = (float*)alloc(MN * DMODEL * 4);
    bf16*  attnb = (bf16*)alloc(MN * DMODEL * 2);
    float* tmp = (float*)alloc(MN * DMODEL * 4);
    float* hf  = (float*)alloc(MN * FFN * 4);
    bf16*  hb  = (bf16*) alloc(MN * FFN * 2);
    bf16*  gb  = (bf16*) alloc(MN * DMODEL * 2);
    float* Ms  = (float*)alloc((size_t)BATCH * NH * LSEQ * 4);
    int*   tk  = (int*)  alloc((size_t)BATCH * NH * NSAMP * 4);
    bf16*  wqb = (bf16*)alloc((size_t)2 * DMODEL * DMODEL * 2);   // all N-major
    bf16*  wkb = (bf16*)alloc((size_t)2 * DMODEL * DMODEL * 2);
    bf16*  wvb = (bf16*)alloc((size_t)2 * DMODEL * DMODEL * 2);
    bf16*  wob = (bf16*)alloc((size_t)2 * DMODEL * DMODEL * 2);
    bf16*  w1b = (bf16*)alloc((size_t)2 * DMODEL * FFN * 2);
    bf16*  w2b = (bf16*)alloc((size_t)2 * FFN * DMODEL * 2);
    bf16*  pwb = (bf16*)alloc((size_t)PROJK * DMODEL * 2);

    // ---- weight conversion + transpose to N-major bf16 (once per call) ----
    for (int i = 0; i < 2; ++i) {
        const size_t oDD = (size_t)i * DMODEL * DMODEL;
        const size_t oDF = (size_t)i * DMODEL * FFN;
        k_f2bf_t<<<dim3(4, 4),  256, 0, stream>>>(Wq + oDD, wqb + oDD, DMODEL, DMODEL);
        k_f2bf_t<<<dim3(4, 4),  256, 0, stream>>>(Wk + oDD, wkb + oDD, DMODEL, DMODEL);
        k_f2bf_t<<<dim3(4, 4),  256, 0, stream>>>(Wv + oDD, wvb + oDD, DMODEL, DMODEL);
        k_f2bf_t<<<dim3(4, 4),  256, 0, stream>>>(Wo + oDD, wob + oDD, DMODEL, DMODEL);
        k_f2bf_t<<<dim3(4, 16), 256, 0, stream>>>(W1 + oDF, w1b + oDF, DMODEL, FFN);
        k_f2bf_t<<<dim3(16, 4), 256, 0, stream>>>(W2 + oDF, w2b + oDF, FFN, DMODEL);
    }
    k_f2bf_t<<<dim3(PROJK / 32, 4), 256, 0, stream>>>(proj_W, pwb, PROJK, DMODEL);

    // ---- embedding ----
    k_embed<<<MNROWS, 128, 0, stream>>>(x_enc, conv_W, x, xb);

    // ---- 2 encoder layers ----
    for (int i = 0; i < 2; ++i) {
        const size_t oDD = (size_t)i * DMODEL * DMODEL;
        const size_t oDF = (size_t)i * DMODEL * FFN;
        const dim3 gP(MNROWS / 128, DMODEL / 64);   // M=32768, N=128
        k_gemm<<<gP, 256, 0, stream>>>(xb, wqb + oDD, bq + i * DMODEL, qf, nullptr,
                                       MNROWS, DMODEL, DMODEL, 0);
        k_gemm<<<gP, 256, 0, stream>>>(xb, wkb + oDD, bk + i * DMODEL, kf, nullptr,
                                       MNROWS, DMODEL, DMODEL, 0);
        k_gemm<<<gP, 256, 0, stream>>>(xb, wvb + oDD, bv + i * DMODEL, vf, nullptr,
                                       MNROWS, DMODEL, DMODEL, 0);

        // ProbSparse attention
        k_scores<<<(BATCH * NH * LSEQ) / 256, 256, 0, stream>>>(qf, kf, Ms);
        k_topk<<<BATCH * NH, 256, 0, stream>>>(Ms, tk);
        k_ctx<<<BATCH * NH, 256, 0, stream>>>(vf, attnb);
        k_topq<<<BATCH * NH * NSAMP, 256, 0, stream>>>(qf, kf, vf, tk, attnb);

        // output projection + residual + LN1
        k_gemm<<<gP, 256, 0, stream>>>(attnb, wob + oDD, bo + i * DMODEL, tmp, nullptr,
                                       MNROWS, DMODEL, DMODEL, 0);
        k_add_ln<<<MNROWS, 128, 0, stream>>>(x, tmp, ln1_g + i * DMODEL,
                                             ln1_b + i * DMODEL, x, xb, 0);

        // FFN: gelu(x@W1+b1)@W2+b2, residual + LN2
        const dim3 gF1(MNROWS / 128, FFN / 64);
        k_gemm<<<gF1, 256, 0, stream>>>(xb, w1b + oDF, b1 + i * FFN, hf, hb,
                                        MNROWS, FFN, DMODEL, 1);
        k_gemm<<<gP, 256, 0, stream>>>(hb, w2b + oDF, b2 + i * DMODEL, tmp, nullptr,
                                       MNROWS, DMODEL, FFN, 0);
        k_add_ln<<<MNROWS, 128, 0, stream>>>(x, tmp, ln2_g + i * DMODEL,
                                             ln2_b + i * DMODEL, x, xb, 0);
    }

    // ---- final LN + GELU (bf16 only), then split-K projection ----
    k_add_ln<<<MNROWS, 128, 0, stream>>>(x, nullptr, norm_g, norm_b,
                                         nullptr, gb, 1);
    k_out_init<<<(BATCH * DMODEL) / 256, 256, 0, stream>>>(proj_b, outp);
    k_proj<<<PROJK / 2048, 128, 0, stream>>>(gb, pwb, outp);
}
